// MultiHeadAttention_11562051961507
// MI455X (gfx1250) — matrix operand order repaired and checked by reading, not compiled, not run on hardware
//
#include <hip/hip_runtime.h>
#include <hip/hip_bf16.h>

typedef __bf16 bf16_t;
typedef __attribute__((ext_vector_type(16))) __bf16 v16bf;
typedef __attribute__((ext_vector_type(8)))  float  v8f;

#define D_MODEL 1024
#define N_HEADS 16
#define D_K     64
#define SEQ     1024
#define BATCH   4
#define NREL    257
#define NP      288            // rel bucket dim padded to 9*32
#define SCALE_F 0.125f         // 1/sqrt(64)

#define WMMA_BF16(A, B, C) \
  __builtin_amdgcn_wmma_f32_16x16x32_bf16(false, (A), false, (B), (short)0, (C), false, false)

static __device__ __forceinline__ v8f zero8() {
  v8f z;
#pragma unroll
  for (int i = 0; i < 8; ++i) z[i] = 0.0f;
  return z;
}

// gfx1250 async copy: 16B global -> LDS per lane, tracked by ASYNCcnt.
static __device__ __forceinline__ void async_copy_b128(void* lds_dst, const void* gsrc) {
  asm volatile("global_load_async_to_lds_b128 %0, %1, off"
               :: "v"((unsigned)(size_t)lds_dst),
                  "v"((unsigned long long)(size_t)gsrc)
               : "memory");
}
static __device__ __forceinline__ void wait_async0() {
  asm volatile("s_wait_asynccnt 0" ::: "memory");
}

// A-fragment: 16x32 bf16, rows m0..m0+15, cols k0..k0+31, LDS row-major (stride lda).
// Layout: lanes 0-15 -> K {0..15}, lanes 16-31 -> K {16..31} (contiguous, matches B loaders).
static __device__ __forceinline__ v16bf load_frag_a(const bf16_t* base, int lda, int m0, int k0) {
  const int lane = threadIdx.x & 31;
  const int m = m0 + (lane & 15);
  const int khalf = (lane < 16) ? 0 : 16;
  v16bf a;
#pragma unroll
  for (int i = 0; i < 8; ++i) {
    const int kb = k0 + 2 * i + khalf;
    a[2 * i]     = base[m * lda + kb];
    a[2 * i + 1] = base[m * lda + kb + 1];
  }
  return a;
}

// B-fragment: 32x16 bf16, B[k][n] = tile[n*ldn + k]  (tile stored N-major).
static __device__ __forceinline__ v16bf load_frag_b_nmajor(const bf16_t* base, int ldn, int k0, int n0) {
  const int lane = threadIdx.x & 31;
  const int n = n0 + (lane & 15);
  const int kb = k0 + ((lane < 16) ? 0 : 16);
  v16bf b;
#pragma unroll
  for (int i = 0; i < 8; ++i) {
    b[2 * i]     = base[n * ldn + kb + 2 * i];
    b[2 * i + 1] = base[n * ldn + kb + 2 * i + 1];
  }
  return b;
}

// B-fragment: 32x16 bf16, B[k][n] = tile[k*ldk + n]  (tile stored K-major).
static __device__ __forceinline__ v16bf load_frag_b_kmajor(const bf16_t* base, int ldk, int k0, int n0) {
  const int lane = threadIdx.x & 31;
  const int n = n0 + (lane & 15);
  const int kb = k0 + ((lane < 16) ? 0 : 16);
  v16bf b;
#pragma unroll
  for (int i = 0; i < 8; ++i) {
    b[2 * i]     = base[(kb + 2 * i) * ldk + n];
    b[2 * i + 1] = base[(kb + 2 * i + 1) * ldk + n];
  }
  return b;
}

// ---------------------------------------------------------------------------
// Kernel 1: projection  out[b,h,s,d] = X @ W^T + bias  (fp32 in, bf16 out)
// Block tile 128x64, 8 waves (4x2). Double-buffered LDS; two-phase staging so
// all global loads are in flight while WMMAs of the previous tile execute.
// ---------------------------------------------------------------------------
__global__ __launch_bounds__(256) void proj_kernel(const float* __restrict__ X,
                                                   const float* __restrict__ W,
                                                   const float* __restrict__ bias,
                                                   bf16_t* __restrict__ out) {
  __shared__ bf16_t sA[2][128 * 40];
  __shared__ bf16_t sB[2][64 * 40];
  const int tid = threadIdx.x;
  const int lane = tid & 31;
  const int wv = tid >> 5, wm = wv >> 1, wn = wv & 1;
  const int n0 = blockIdx.x * 64;
  const int m0 = blockIdx.y * 128;
  const int hi = lane >> 4, col = lane & 15;

  float4 fa[4], fb[2];
  auto load_regs = [&](int k0) {
#pragma unroll
    for (int i = 0; i < 4; ++i) {
      const int t = tid + (i << 8);
      const int r = t >> 3, kq = (t & 7) << 2;
      fa[i] = *(const float4*)(X + (size_t)(m0 + r) * D_MODEL + k0 + kq);
    }
#pragma unroll
    for (int i = 0; i < 2; ++i) {
      const int t = tid + (i << 8);
      const int n = t >> 3, kq = (t & 7) << 2;
      fb[i] = *(const float4*)(W + (size_t)(n0 + n) * D_MODEL + k0 + kq);
    }
  };
  auto store_regs = [&](int p) {
#pragma unroll
    for (int i = 0; i < 4; ++i) {
      const int t = tid + (i << 8);
      const int r = t >> 3, kq = (t & 7) << 2;
      bf16_t* dst = &sA[p][r * 40 + kq];
      dst[0] = (bf16_t)fa[i].x; dst[1] = (bf16_t)fa[i].y;
      dst[2] = (bf16_t)fa[i].z; dst[3] = (bf16_t)fa[i].w;
    }
#pragma unroll
    for (int i = 0; i < 2; ++i) {
      const int t = tid + (i << 8);
      const int n = t >> 3, kq = (t & 7) << 2;
      bf16_t* dst = &sB[p][n * 40 + kq];
      dst[0] = (bf16_t)fb[i].x; dst[1] = (bf16_t)fb[i].y;
      dst[2] = (bf16_t)fb[i].z; dst[3] = (bf16_t)fb[i].w;
    }
  };

  v8f acc[2][2];
#pragma unroll
  for (int i = 0; i < 2; ++i)
#pragma unroll
    for (int j = 0; j < 2; ++j) acc[i][j] = zero8();

  load_regs(0);
  int p = 0;
  for (int k0 = 0; k0 < D_MODEL; k0 += 32) {
    store_regs(p);
    __syncthreads();
    if (k0 + 32 < D_MODEL) load_regs(k0 + 32);   // overlaps with WMMA below
    const v16bf a0 = load_frag_a(sA[p], 40, wm * 32, 0);
    const v16bf a1 = load_frag_a(sA[p], 40, wm * 32 + 16, 0);
    const v16bf b0 = load_frag_b_nmajor(sB[p], 40, 0, wn * 32);
    const v16bf b1 = load_frag_b_nmajor(sB[p], 40, 0, wn * 32 + 16);
    acc[0][0] = WMMA_BF16(a0, b0, acc[0][0]);
    acc[0][1] = WMMA_BF16(a0, b1, acc[0][1]);
    acc[1][0] = WMMA_BF16(a1, b0, acc[1][0]);
    acc[1][1] = WMMA_BF16(a1, b1, acc[1][1]);
    p ^= 1;
  }

#pragma unroll
  for (int mi = 0; mi < 2; ++mi)
#pragma unroll
    for (int ni = 0; ni < 2; ++ni)
#pragma unroll
      for (int v = 0; v < 8; ++v) {
        const int rowg = m0 + wm * 32 + mi * 16 + v + 8 * hi;
        const int colg = n0 + wn * 32 + ni * 16 + col;
        const float val = acc[mi][ni][v] + bias[colg];
        const int bb = rowg >> 10, ss = rowg & 1023, hh = colg >> 6, dd = colg & 63;
        out[((size_t)(bb * N_HEADS + hh) * SEQ + ss) * D_K + dd] = (bf16_t)val;
      }
}

// ---------------------------------------------------------------------------
// Kernel 2: Qrel[R, p] = sum_d Q[R, d] * rel_k[p, d]   (R = (b*16+h)*1024 + l)
// ---------------------------------------------------------------------------
__global__ __launch_bounds__(128) void qrel_kernel(const bf16_t* __restrict__ Qbf,
                                                   const float* __restrict__ relk,
                                                   bf16_t* __restrict__ Qrl) {
  __shared__ bf16_t sQ[64 * 64];
  __shared__ bf16_t sRK[NP * 64];
  const int tid = threadIdx.x, lane = tid & 31, wv = tid >> 5;   // 4 waves
  const size_t R0 = (size_t)blockIdx.x * 64;
  const int hi = lane >> 4, col = lane & 15;

  for (int t = tid; t < 512; t += 128)                 // Q tile 64x64: async copy
    async_copy_b128(&sQ[(t >> 3) * 64 + ((t & 7) << 3)],
                    Qbf + (R0 + (t >> 3)) * D_K + ((t & 7) << 3));

  for (int t = tid; t < NP * 16; t += 128) {           // rel_k 288x64 (guarded cvt)
    const int pp = t >> 4, kq = (t & 15) << 2;
    float4 f = make_float4(0.f, 0.f, 0.f, 0.f);
    if (pp < NREL) f = *(const float4*)(relk + (size_t)pp * D_K + kq);
    bf16_t* dst = &sRK[pp * 64 + kq];
    dst[0] = (bf16_t)f.x; dst[1] = (bf16_t)f.y; dst[2] = (bf16_t)f.z; dst[3] = (bf16_t)f.w;
  }
  wait_async0();
  __syncthreads();

  const v16bf a0 = load_frag_a(sQ, 64, wv * 16, 0);
  const v16bf a1 = load_frag_a(sQ, 64, wv * 16, 32);
#pragma unroll
  for (int nt = 0; nt < NP / 16; ++nt) {
    const v16bf b0 = load_frag_b_nmajor(sRK, 64, 0, nt * 16);
    const v16bf b1 = load_frag_b_nmajor(sRK, 64, 32, nt * 16);
    v8f c = zero8();
    c = WMMA_BF16(a0, b0, c);
    c = WMMA_BF16(a1, b1, c);
#pragma unroll
    for (int v = 0; v < 8; ++v) {
      const size_t rowR = R0 + wv * 16 + v + 8 * hi;
      Qrl[rowR * NP + nt * 16 + col] = (bf16_t)c[v];
    }
  }
}

// ---------------------------------------------------------------------------
// Kernel 3: streaming attention. Block = 2 waves, 32 query rows.
// ---------------------------------------------------------------------------
static __device__ __forceinline__ void score_tiles(const bf16_t* sQ, int mrow0,
                                                   const bf16_t* sK,
                                                   const bf16_t* qrelW,
                                                   const int* maskb,
                                                   int lbase, int j0, v8f sc[4]) {
  const int lane = threadIdx.x & 31;
  const int hi = lane >> 4, col = lane & 15;
  const v16bf a0 = load_frag_a(sQ, 64, mrow0, 0);
  const v16bf a1 = load_frag_a(sQ, 64, mrow0, 32);
#pragma unroll
  for (int nt = 0; nt < 4; ++nt) {
    const v16bf b0 = load_frag_b_nmajor(sK, 64, 0, nt * 16);
    const v16bf b1 = load_frag_b_nmajor(sK, 64, 32, nt * 16);
    v8f c = zero8();
    c = WMMA_BF16(a0, b0, c);
    c = WMMA_BF16(a1, b1, c);
    const int r = j0 + nt * 16 + col;
    const bool mok = maskb[r] != 0;
#pragma unroll
    for (int v = 0; v < 8; ++v) {
      const int row = v + 8 * hi;
      const int l = lbase + row;
      int d = r - l;
      d = d < -128 ? -128 : (d > 128 ? 128 : d);
      const float rel = (float)qrelW[row * NP + d + 128];
      const float s = c[v] * SCALE_F + rel;
      c[v] = mok ? s : -1.0e9f;
    }
    sc[nt] = c;
  }
}

__global__ __launch_bounds__(64) void attn_kernel(const bf16_t* __restrict__ Qbf,
                                                  const bf16_t* __restrict__ Kbf,
                                                  const bf16_t* __restrict__ Vbf,
                                                  const bf16_t* __restrict__ Qrl,
                                                  const int* __restrict__ mask,
                                                  const float* __restrict__ relv,
                                                  bf16_t* __restrict__ ctx) {
  __shared__ bf16_t sQ[32 * 64];
  __shared__ bf16_t sK[64 * 64];
  __shared__ bf16_t sV[64 * 64];
  __shared__ bf16_t sP[2 * 16 * 64];
  __shared__ float  sW[32 * NP];
  __shared__ float  rowMax[32];
  __shared__ float  rowDen[32];

  const int tid = threadIdx.x, lane = tid & 31, wv = tid >> 5;   // 2 waves
  const int bh = blockIdx.y;
  const int b = bh >> 4, h = bh & 15;
  const int l0 = blockIdx.x * 32;
  const size_t Rtile = (size_t)bh * SEQ + l0;
  const int hi = lane >> 4, col = lane & 15;
  const int lbase = l0 + wv * 16;
  const bf16_t* qrelW = Qrl + (Rtile + (size_t)wv * 16) * NP;
  const int* maskb = mask + b * SEQ;
  bf16_t* sPw = sP + wv * 16 * 64;
  const size_t KVbase = (size_t)bh * SEQ * D_K;

  for (int t = tid; t < 256; t += 64)                   // stage Q tile (async)
    async_copy_b128(&sQ[(t >> 3) * 64 + ((t & 7) << 3)],
                    Qbf + (Rtile + (t >> 3)) * D_K + ((t & 7) << 3));
  for (int t = tid; t < 32 * NP; t += 64) sW[t] = 0.0f;
  if (tid < 32) { rowMax[tid] = -3.0e38f; rowDen[tid] = 0.0f; }
  wait_async0();
  __syncthreads();

  // ---------------- pass 1: global row maxima ----------------
  for (int j0 = 0; j0 < SEQ; j0 += 64) {
    for (int t = tid; t < 512; t += 64)
      async_copy_b128(&sK[(t >> 3) * 64 + ((t & 7) << 3)],
                      Kbf + KVbase + (size_t)(j0 + (t >> 3)) * D_K + ((t & 7) << 3));
    if (j0 + 64 < SEQ) __builtin_prefetch(Kbf + KVbase + (size_t)(j0 + 64) * D_K, 0, 0);
    wait_async0();
    __syncthreads();
    v8f sc[4];
    score_tiles(sQ, wv * 16, sK, qrelW, maskb, lbase, j0, sc);
#pragma unroll
    for (int v = 0; v < 8; ++v) {
      float m = fmaxf(fmaxf(sc[0][v], sc[1][v]), fmaxf(sc[2][v], sc[3][v]));
#pragma unroll
      for (int off = 1; off < 16; off <<= 1) m = fmaxf(m, __shfl_xor(m, off, 32));
      if (col == 0) {
        const int row = wv * 16 + v + 8 * hi;
        rowMax[row] = fmaxf(rowMax[row], m);
      }
    }
    __syncthreads();
  }

  // ---------------- pass 2: exp, denom, O += P@V, bucket scatter ----------------
  v8f oacc[4];
#pragma unroll
  for (int nt = 0; nt < 4; ++nt) oacc[nt] = zero8();

  for (int j0 = 0; j0 < SEQ; j0 += 64) {
    for (int t = tid; t < 512; t += 64) {
      const int r = t >> 3, kq = (t & 7) << 3;
      async_copy_b128(&sK[r * 64 + kq], Kbf + KVbase + (size_t)(j0 + r) * D_K + kq);
      async_copy_b128(&sV[r * 64 + kq], Vbf + KVbase + (size_t)(j0 + r) * D_K + kq);
    }
    wait_async0();
    __syncthreads();
    v8f sc[4];
    score_tiles(sQ, wv * 16, sK, qrelW, maskb, lbase, j0, sc);

    float rowM[8];
#pragma unroll
    for (int v = 0; v < 8; ++v) rowM[v] = rowMax[wv * 16 + v + 8 * hi];
    float dsum[8] = {};

#pragma unroll
    for (int nt = 0; nt < 4; ++nt) {
      const int r = j0 + nt * 16 + col;
#pragma unroll
      for (int v = 0; v < 8; ++v) {
        const int row = v + 8 * hi;
        const int l = lbase + row;
        const float pe = __expf(sc[nt][v] - rowM[v]);
        dsum[v] += pe;
        int d = r - l;
        d = d < -128 ? -128 : (d > 128 ? 128 : d);
        atomicAdd(&sW[(wv * 16 + row) * NP + d + 128], pe);
        sPw[row * 64 + nt * 16 + col] = (bf16_t)pe;
      }
    }
#pragma unroll
    for (int v = 0; v < 8; ++v) {
      float s = dsum[v];
#pragma unroll
      for (int off = 1; off < 16; off <<= 1) s += __shfl_xor(s, off, 32);
      if (col == 0) rowDen[wv * 16 + v + 8 * hi] += s;
    }
    __syncthreads();
#pragma unroll
    for (int kc = 0; kc < 2; ++kc) {
      const v16bf ap = load_frag_a(sPw, 64, 0, kc * 32);
#pragma unroll
      for (int nt = 0; nt < 4; ++nt) {
        const v16bf bv = load_frag_b_kmajor(sV, 64, kc * 32, nt * 16);
        oacc[nt] = WMMA_BF16(ap, bv, oacc[nt]);
      }
    }
    __syncthreads();
  }

  // ---------------- rel_v: racc = W(32x288) @ rel_v(288x64) ----------------
  v8f racc[4];
#pragma unroll
  for (int nt = 0; nt < 4; ++nt) racc[nt] = zero8();
  const int khalf = hi ? 16 : 0;
#pragma unroll
  for (int kc = 0; kc < NP / 32; ++kc) {
    const int kk = kc * 32;
    v16bf aw;
    const int m = wv * 16 + col;
#pragma unroll
    for (int i = 0; i < 8; ++i) {
      const int kb = kk + 2 * i + khalf;
      aw[2 * i]     = (bf16_t)sW[m * NP + kb];
      aw[2 * i + 1] = (bf16_t)sW[m * NP + kb + 1];
    }
#pragma unroll
    for (int nt = 0; nt < 4; ++nt) {
      v16bf bw;
      const int n = nt * 16 + col;
      const int kb0 = kk + (hi ? 16 : 0);
#pragma unroll
      for (int i = 0; i < 8; ++i) {
        const int p0 = kb0 + 2 * i;
        bw[2 * i]     = (p0     < NREL) ? (bf16_t)relv[(size_t)p0 * D_K + n]       : (bf16_t)0.0f;
        bw[2 * i + 1] = (p0 + 1 < NREL) ? (bf16_t)relv[(size_t)(p0 + 1) * D_K + n] : (bf16_t)0.0f;
      }
      racc[nt] = WMMA_BF16(aw, bw, racc[nt]);
    }
  }

  // ---------------- normalize and store ctx [b, s, h, d] ----------------
#pragma unroll
  for (int v = 0; v < 8; ++v) {
    const int row = wv * 16 + v + 8 * hi;
    const float invd = 1.0f / rowDen[row];
    const int l = l0 + row;
#pragma unroll
    for (int nt = 0; nt < 4; ++nt) {
      const int dg = nt * 16 + col;
      const float val = (oacc[nt][v] + racc[nt][v]) * invd;
      ctx[(((size_t)(b * SEQ + l)) * N_HEADS + h) * D_K + dg] = (bf16_t)val;
    }
  }
}

// ---------------------------------------------------------------------------
// Kernel 4: out = ctx @ w_o^T + b_o   (bf16 in, fp32 out, row-major)
// A tile staged with gfx1250 async copies; B tile via two-phase registers.
// Double-buffered LDS.
// ---------------------------------------------------------------------------
__global__ __launch_bounds__(256) void outproj_kernel(const bf16_t* __restrict__ X,
                                                      const float* __restrict__ W,
                                                      const float* __restrict__ bias,
                                                      float* __restrict__ out) {
  __shared__ bf16_t sA[2][128 * 40];
  __shared__ bf16_t sB[2][64 * 40];
  const int tid = threadIdx.x;
  const int lane = tid & 31;
  const int wv = tid >> 5, wm = wv >> 1, wn = wv & 1;
  const int n0 = blockIdx.x * 64;
  const int m0 = blockIdx.y * 128;
  const int hi = lane >> 4, col = lane & 15;

  float4 fb[2];
  auto issueA = [&](int k0, int p) {
#pragma unroll
    for (int i = 0; i < 2; ++i) {
      const int t = tid + (i << 8);
      const int r = t >> 2, kq = (t & 3) << 3;
      async_copy_b128(&sA[p][r * 40 + kq], X + (size_t)(m0 + r) * D_MODEL + k0 + kq);
    }
  };
  auto loadB = [&](int k0) {
#pragma unroll
    for (int i = 0; i < 2; ++i) {
      const int t = tid + (i << 8);
      const int n = t >> 3, kq = (t & 7) << 2;
      fb[i] = *(const float4*)(W + (size_t)(n0 + n) * D_MODEL + k0 + kq);
    }
  };
  auto storeB = [&](int p) {
#pragma unroll
    for (int i = 0; i < 2; ++i) {
      const int t = tid + (i << 8);
      const int n = t >> 3, kq = (t & 7) << 2;
      bf16_t* dst = &sB[p][n * 40 + kq];
      dst[0] = (bf16_t)fb[i].x; dst[1] = (bf16_t)fb[i].y;
      dst[2] = (bf16_t)fb[i].z; dst[3] = (bf16_t)fb[i].w;
    }
  };

  v8f acc[2][2];
#pragma unroll
  for (int i = 0; i < 2; ++i)
#pragma unroll
    for (int j = 0; j < 2; ++j) acc[i][j] = zero8();

  issueA(0, 0);
  loadB(0);
  int p = 0;
  for (int k0 = 0; k0 < D_MODEL; k0 += 32) {
    storeB(p);
    wait_async0();
    __syncthreads();
    if (k0 + 32 < D_MODEL) {
      issueA(k0 + 32, p ^ 1);     // async fill of next buffer overlaps WMMA
      loadB(k0 + 32);
    }
    const v16bf a0 = load_frag_a(sA[p], 40, wm * 32, 0);
    const v16bf a1 = load_frag_a(sA[p], 40, wm * 32 + 16, 0);
    const v16bf b0 = load_frag_b_nmajor(sB[p], 40, 0, wn * 32);
    const v16bf b1 = load_frag_b_nmajor(sB[p], 40, 0, wn * 32 + 16);
    acc[0][0] = WMMA_BF16(a0, b0, acc[0][0]);
    acc[0][1] = WMMA_BF16(a0, b1, acc[0][1]);
    acc[1][0] = WMMA_BF16(a1, b0, acc[1][0]);
    acc[1][1] = WMMA_BF16(a1, b1, acc[1][1]);
    p ^= 1;
  }

#pragma unroll
  for (int mi = 0; mi < 2; ++mi)
#pragma unroll
    for (int ni = 0; ni < 2; ++ni)
#pragma unroll
      for (int v = 0; v < 8; ++v) {
        const int rowg = m0 + wm * 32 + mi * 16 + v + 8 * hi;
        const int colg = n0 + wn * 32 + ni * 16 + col;
        out[(size_t)rowg * D_MODEL + colg] = acc[mi][ni][v] + bias[colg];
      }
}

// ---------------------------------------------------------------------------
extern "C" void kernel_launch(void* const* d_in, const int* in_sizes, int n_in,
                              void* d_out, int out_size, void* d_ws, size_t ws_size,
                              hipStream_t stream) {
  const float* query = (const float*)d_in[0];
  const float* key   = (const float*)d_in[1];
  const float* value = (const float*)d_in[2];
  const int*   mask  = (const int*)d_in[3];
  const float* w_q = (const float*)d_in[4];
  const float* b_q = (const float*)d_in[5];
  const float* w_k = (const float*)d_in[6];
  const float* b_k = (const float*)d_in[7];
  const float* w_v = (const float*)d_in[8];
  const float* b_v = (const float*)d_in[9];
  const float* w_o = (const float*)d_in[10];
  const float* b_o = (const float*)d_in[11];
  const float* rel_k = (const float*)d_in[12];
  const float* rel_v = (const float*)d_in[13];

  const size_t nQKV = (size_t)BATCH * N_HEADS * SEQ * D_K;   // 4 Mi elems
  bf16_t* Qbf = (bf16_t*)d_ws;
  bf16_t* Kbf = Qbf + nQKV;
  bf16_t* Vbf = Kbf + nQKV;
  bf16_t* Qrl = Vbf + nQKV;                                  // 65536 x NP
  bf16_t* ctx = Qrl + (size_t)BATCH * N_HEADS * SEQ * NP;

  const dim3 gProj(D_MODEL / 64, (BATCH * SEQ) / 128);
  proj_kernel<<<gProj, 256, 0, stream>>>(query, w_q, b_q, Qbf);
  proj_kernel<<<gProj, 256, 0, stream>>>(key,   w_k, b_k, Kbf);
  proj_kernel<<<gProj, 256, 0, stream>>>(value, w_v, b_v, Vbf);

  qrel_kernel<<<(BATCH * N_HEADS * SEQ) / 64, 128, 0, stream>>>(Qbf, rel_k, Qrl);

  attn_kernel<<<dim3(SEQ / 32, BATCH * N_HEADS), 64, 0, stream>>>(
      Qbf, Kbf, Vbf, Qrl, mask, rel_v, ctx);

  outproj_kernel<<<gProj, 256, 0, stream>>>(ctx, w_o, b_o, (float*)d_out);
}
